// W8A8OF16LinearDynamicInputScale_9878424781510
// MI455X (gfx1250) — 8-bit matrix operand order repaired and checked by reading, compiled, not run on hardware
//
#include <hip/hip_runtime.h>
#include <hip/hip_bf16.h>
#include <stdint.h>


typedef int v8i __attribute__((ext_vector_type(8)));
typedef int v4i __attribute__((ext_vector_type(4)));
typedef int v2i __attribute__((ext_vector_type(2)));

#define N_TOK 4096
#define K_DIM 4096
#define M_DIM 12288

#define BN 128
#define BM 128
#define BK 128
#define STR 144

__global__ __launch_bounds__(256) void pack_i32_to_i8(
    const int* __restrict__ in, unsigned int* __restrict__ out, int n4)
{
    int i = blockIdx.x * blockDim.x + threadIdx.x;
    if (i >= n4) return;
    const int4 v = reinterpret_cast<const int4*>(in)[i];
    unsigned int p = (v.x & 0xFF)
                   | ((v.y & 0xFF) << 8)
                   | ((v.z & 0xFF) << 16)
                   | (((unsigned int)v.w) << 24);
    out[i] = p;
}

__global__ __launch_bounds__(256) void w8a8_gemm_wmma(
    const signed char* __restrict__ x8,
    const signed char* __restrict__ w8,
    const float* __restrict__ input_scale,
    const float* __restrict__ dequant_scale,
    const float* __restrict__ bias,
    float* __restrict__ out)
{
    __shared__ __align__(16) unsigned char smem[(BN + BM) * STR];
    unsigned char* sA = smem;
    unsigned char* sB = smem + BN * STR;

    const int tid  = threadIdx.x;
    const int lane = tid & 31;
    const int wave = tid >> 5;
    const int waveN = (wave >> 2) * 64;
    const int waveM = (wave & 3) * 32;

    const int m0 = blockIdx.x * BM;
    const int n0 = blockIdx.y * BN;

    const int lm = lane & 15;
    const int lh = lane >> 4;

    const unsigned char* aBase = sA + (waveN + lm) * STR + lh * 16;
    const unsigned char* bBase = sB + (waveM + lm) * STR + lh * 16;

    v8i acc[4][2];
#pragma unroll
    for (int sn = 0; sn < 4; ++sn)
#pragma unroll
        for (int sm = 0; sm < 2; ++sm)
            acc[sn][sm] = v8i{0, 0, 0, 0, 0, 0, 0, 0};

    for (int kb = 0; kb < K_DIM; kb += BK) {
        __syncthreads();

#pragma unroll
        for (int i = 0; i < 4; ++i) {
            const int idx = tid + i * 256;
            const int row = idx >> 3;
            const int q   = idx & 7;
            const uint4 av = *reinterpret_cast<const uint4*>(
                x8 + (size_t)(n0 + row) * K_DIM + kb + q * 16);
            *reinterpret_cast<uint4*>(sA + row * STR + q * 16) = av;
            const uint4 bv = *reinterpret_cast<const uint4*>(
                w8 + (size_t)(m0 + row) * K_DIM + kb + q * 16);
            *reinterpret_cast<uint4*>(sB + row * STR + q * 16) = bv;
        }
        __syncthreads();

#pragma unroll
        for (int kc = 0; kc < 2; ++kc) {
            const int ko = kc * 64;

            v8i bfrag[2];
#pragma unroll
            for (int sm = 0; sm < 2; ++sm) {
                const unsigned char* bp = bBase + sm * (16 * STR) + ko;
                const v4i lo = *reinterpret_cast<const v4i*>(bp);
                const v4i hi = *reinterpret_cast<const v4i*>(bp + 32);
                v8i f;
#pragma unroll
                for (int j = 0; j < 4; ++j) { f[j] = lo[j]; f[4 + j] = hi[j]; }
                bfrag[sm] = f;
            }

            v8i afrag[4];
#pragma unroll
            for (int sn = 0; sn < 4; ++sn) {
                const unsigned char* ap = aBase + sn * (16 * STR) + ko;
                v8i a;
#pragma unroll
                for (int c = 0; c < 4; ++c) {
                    const v2i d = *reinterpret_cast<const v2i*>(ap + (c & 1) * 8 + (c >> 1) * 32);
                    a[2 * c]     = d[0];
                    a[2 * c + 1] = d[1];
                }
                afrag[sn] = a;
            }

#pragma unroll
            for (int sn = 0; sn < 4; ++sn)
#pragma unroll
                for (int sm = 0; sm < 2; ++sm)
                    acc[sn][sm] = __builtin_amdgcn_wmma_i32_16x16x64_iu8(
                         true, afrag[sn],
                         true, bfrag[sm],
                        acc[sn][sm],
                         false,  false);
        }
    }

    float iscale[4][8];
#pragma unroll
    for (int sn = 0; sn < 4; ++sn) {
        const int rowbase = n0 + waveN + sn * 16 + lh * 8;
#pragma unroll
        for (int r = 0; r < 8; ++r)
            iscale[sn][r] = input_scale[rowbase + r];
    }

#pragma unroll
    for (int sm = 0; sm < 2; ++sm) {
        const int col = m0 + waveM + sm * 16 + lm;
        const float dsc = dequant_scale[col];
        const float bi  = bias[col];
#pragma unroll
        for (int sn = 0; sn < 4; ++sn) {
            const int rowbase = n0 + waveN + sn * 16 + lh * 8;
#pragma unroll
            for (int r = 0; r < 8; ++r) {
                const float v = (float)acc[sn][sm][r] * iscale[sn][r] * dsc + bi;
                out[(size_t)(rowbase + r) * M_DIM + col] = v;
            }
        }
    }
}

extern "C" void kernel_launch(void* const* d_in, const int* in_sizes, int n_in,
                              void* d_out, int out_size, void* d_ws, size_t ws_size,
                              hipStream_t stream)
{
    const int*   x  = (const int*)d_in[0];
    const int*   w  = (const int*)d_in[1];
    const float* in_scale = (const float*)d_in[2];
    const float* dq_scale = (const float*)d_in[3];
    const float* bias     = (const float*)d_in[4];
    float* out = (float*)d_out;

    unsigned char* x8 = (unsigned char*)d_ws;
    unsigned char* w8 = x8 + (size_t)N_TOK * K_DIM;

    {
        const int n4 = N_TOK * K_DIM / 4;
        pack_i32_to_i8<<<(n4 + 255) / 256, 256, 0, stream>>>(x, (unsigned int*)x8, n4);
    }
    {
        const int n4 = M_DIM * K_DIM / 4;
        pack_i32_to_i8<<<(n4 + 255) / 256, 256, 0, stream>>>(w, (unsigned int*)w8, n4);
    }

    dim3 grid(M_DIM / BM, N_TOK / BN);
    w8a8_gemm_wmma<<<grid, 256, 0, stream>>>(
        (const signed char*)x8, (const signed char*)w8,
        in_scale, dq_scale, bias, out);
}
